// NativeSparseCrossAttention_14224931685014
// MI455X (gfx1250) — matrix operand order repaired and checked by reading, not compiled, not run on hardware
//
#include <hip/hip_runtime.h>
#include <hip/hip_bf16.h>
#include <math.h>

// ---------------------------------------------------------------------------
// NativeSparseCrossAttention for MI455X (gfx1250, wave32, WMMA)
//
// Key optimization: instead of projecting gathered ctx per latent
// (137 GFLOP), precompute K = bytes@Wk and V = bytes@Wv once over all 32768
// token rows (17 GFLOP) in bf16 via v_wmma_f32_16x16x32_bf16. K+V in bf16 is
// 67 MB -> resident in the 192MB L2, so the per-latent gather phase
// (~537 MB of reads) is served from L2, and HBM traffic is ~100 MB total.
//
// GEMM waves keep 4 accumulators (16x64 strip) to reuse each A fragment 4x:
// 2.5 b128 loads per WMMA instead of 4.
// ---------------------------------------------------------------------------

typedef __attribute__((ext_vector_type(16))) __bf16 v16bf;
typedef __attribute__((ext_vector_type(8)))  __bf16 v8bf;
typedef __attribute__((ext_vector_type(2)))  __bf16 v2bf;
typedef __attribute__((ext_vector_type(8)))  float  v8f;

#define B_     2
#define N_L    128
#define D_     512
#define N_T    16384
#define D_IN   256
#define NB     256
#define BS     64
#define TOPK   16
#define NH     8
#define HD     64
#define RDIM   64
#define MROWS  (B_ * N_T)          // 32768 rows into the big GEMMs
#define NROWS  (B_ * N_L)          // 256 latent rows

// ---------------------------------------------------------------------------
// Repack byte_array (fp32 [32768,256]) into bf16 WMMA A-fragments.
// ISA 16-bit A 16x32 layout: lane L holds M=L&15;
//   k(j) = kt*32 + ((L>=16)?8:0) + ((j>=8)?16:0) + (j&7)
// Fragment storage: Abuf[((mt*8 + kt)*32 + lane)*16 + j]  == Abuf[tid*16+j]
// ---------------------------------------------------------------------------
__global__ __launch_bounds__(256) void nsca_prep_A(
    const float* __restrict__ X, __bf16* __restrict__ Abuf) {
  int tid = blockIdx.x * 256 + threadIdx.x;           // 524288 total
  int lane = tid & 31;
  int kt   = (tid >> 5) & 7;
  int mt   = tid >> 8;
  int m     = mt * 16 + (lane & 15);
  int kbase = kt * 32 + ((lane & 16) ? 8 : 0);
  const float* src = X + (size_t)m * D_IN + kbase;
  __bf16* dst = Abuf + (size_t)tid * 16;
#pragma unroll
  for (int j = 0; j < 8; j++) dst[j]     = (__bf16)src[j];       // K = kbase+0..7
#pragma unroll
  for (int j = 0; j < 8; j++) dst[8 + j] = (__bf16)src[16 + j];  // K = kbase+16..23
}

// ---------------------------------------------------------------------------
// Repack W (fp32 [256,512]) into bf16 WMMA B-fragments (32x16 tile):
// lane L holds N = nt*16 + (L&15); K = kt*32 + ((L>=16)?8:0) + ((j>=8)?16:0) + (j&7)
// ---------------------------------------------------------------------------
__global__ __launch_bounds__(256) void nsca_prep_B(
    const float* __restrict__ W, __bf16* __restrict__ Bf) {
  int tid = blockIdx.x * 256 + threadIdx.x;           // 8192 total
  int lane = tid & 31;
  int kt   = (tid >> 5) & 7;
  int nt   = tid >> 8;
  int n     = nt * 16 + (lane & 15);
  int kbase = kt * 32 + ((lane & 16) ? 8 : 0);
  __bf16* dst = Bf + (size_t)tid * 16;
#pragma unroll
  for (int j = 0; j < 16; j++)
    dst[j] = (__bf16)W[(size_t)(kbase + (j & 7) + ((j >= 8) ? 16 : 0)) * D_ + n];
}

// ---------------------------------------------------------------------------
// Big GEMM: C[32768,512](bf16) = A[32768,256] @ W[256,512], via WMMA bf16.
// One workgroup per 16-row strip (mt); wave wi covers ntiles wi*4..wi*4+3
// with 4 accumulators, reusing each A fragment 4x. All 8 waves share the
// same A strip addresses -> L0 hits.
// ---------------------------------------------------------------------------
#define WMMA_BF16(A, Bv, Cv) \
  __builtin_amdgcn_wmma_f32_16x16x32_bf16(false, (A), false, (Bv), (short)0, (Cv), false, false)

__global__ __launch_bounds__(256) void nsca_wmma_gemm(
    const __bf16* __restrict__ Afrag, const __bf16* __restrict__ Bfrag,
    __bf16* __restrict__ C) {
  int wi   = threadIdx.x >> 5;                // 0..7 -> n-group
  int lane = threadIdx.x & 31;
  int mt   = blockIdx.x;                      // 2048 m-tiles
  int nt0  = wi * 4;                          // 4 consecutive n-tiles
  const __bf16* Ap = Afrag + (size_t)mt * 8 * 512 + lane * 16;
  const __bf16* Bp = Bfrag + (size_t)nt0 * 8 * 512 + lane * 16;
  v8f acc0 = {}, acc1 = {}, acc2 = {}, acc3 = {};
#pragma unroll
  for (int kt = 0; kt < 8; kt++) {
    v16bf a  = *(const v16bf*)(Ap + kt * 512);
    v16bf b0 = *(const v16bf*)(Bp + kt * 512);
    v16bf b1 = *(const v16bf*)(Bp + 1 * 4096 + kt * 512);
    v16bf b2 = *(const v16bf*)(Bp + 2 * 4096 + kt * 512);
    v16bf b3 = *(const v16bf*)(Bp + 3 * 4096 + kt * 512);
    acc0 = WMMA_BF16(a, b0, acc0);
    acc1 = WMMA_BF16(a, b1, acc1);
    acc2 = WMMA_BF16(a, b2, acc2);
    acc3 = WMMA_BF16(a, b3, acc3);
  }
  // D layout: lane holds N = nt*16+(lane&15); VGPR r -> M = r + 8*(lane>=16)
  int m0 = mt * 16 + ((lane >> 4) << 3);
  int n  = nt0 * 16 + (lane & 15);
#pragma unroll
  for (int r = 0; r < 8; r++) {
    __bf16* crow = C + (size_t)(m0 + r) * D_ + n;
    crow[0]  = (__bf16)acc0[r];
    crow[16] = (__bf16)acc1[r];
    crow[32] = (__bf16)acc2[r];
    crow[48] = (__bf16)acc3[r];
  }
}

// ---------------------------------------------------------------------------
// Block summaries: mean over 64 rows. grid = 512 blocks, 256 threads (=D_IN).
// ---------------------------------------------------------------------------
__global__ __launch_bounds__(256) void nsca_summaries(
    const float* __restrict__ X, float* __restrict__ S) {
  int blk = blockIdx.x;
  int c   = threadIdx.x;
  const float* p = X + (size_t)blk * BS * D_IN + c;
  float acc = 0.f;
#pragma unroll 8
  for (int r = 0; r < BS; r++) acc += p[r * D_IN];
  S[(size_t)blk * D_IN + c] = acc * (1.f / 64.f);
}

// ---------------------------------------------------------------------------
// Small fp32 GEMM (router / q / out projections, all <=0.14 GFLOP).
// C[m,n] = sum_k A[m,k] * (transB ? B[n,k] : B[k,n]) + bias[n]
// ---------------------------------------------------------------------------
__global__ __launch_bounds__(256) void nsca_sgemm(
    const float* __restrict__ A, const float* __restrict__ Bm,
    const float* __restrict__ bias, float* __restrict__ C,
    int M, int N, int Kd, int transB) {
  int t = blockIdx.x * 256 + threadIdx.x;
  if (t >= M * N) return;
  int m = t / N, n = t % N;
  const float* a = A + (size_t)m * Kd;
  float acc = bias ? bias[n] : 0.f;
  if (transB) {
    const float* br = Bm + (size_t)n * Kd;
    for (int k = 0; k < Kd; k++) acc += a[k] * br[k];
  } else {
    for (int k = 0; k < Kd; k++) acc += a[k] * Bm[(size_t)k * N + n];
  }
  C[t] = acc;
}

// ---------------------------------------------------------------------------
// Top-16 of 256 logits per latent row (matches lax.top_k order: value desc,
// earlier index wins ties). One thread per row.
// ---------------------------------------------------------------------------
__global__ __launch_bounds__(256) void nsca_topk(
    const float* __restrict__ logits, int* __restrict__ idx) {
  int r = blockIdx.x * blockDim.x + threadIdx.x;
  if (r >= NROWS) return;
  const float* row = logits + (size_t)r * NB;
  float vals[TOPK];
  int   ids[TOPK];
#pragma unroll
  for (int j = 0; j < TOPK; j++) { vals[j] = -INFINITY; ids[j] = 0; }
  for (int c = 0; c < NB; c++) {
    float v = row[c];
    if (v > vals[TOPK - 1]) {
      int p = TOPK - 1;
      while (p > 0 && v > vals[p - 1]) {
        vals[p] = vals[p - 1]; ids[p] = ids[p - 1]; --p;
      }
      vals[p] = v; ids[p] = c;
    }
  }
  for (int j = 0; j < TOPK; j++) idx[r * TOPK + j] = ids[j];
}

// ---------------------------------------------------------------------------
// Attention: one WG (8 wave32 = 8 heads) per latent row. S = 1024 gathered
// ctx rows (16 blocks x 64). K/V are bf16, L2-resident.
// ---------------------------------------------------------------------------
__global__ __launch_bounds__(256) void nsca_attention(
    const float* __restrict__ qbuf, const __bf16* __restrict__ Kf,
    const __bf16* __restrict__ Vf, const int* __restrict__ idx,
    float* __restrict__ obuf) {
  __shared__ float qs[D_];
  __shared__ float sc[NH * 1024];
  __shared__ int   bidx[TOPK];

  int r = blockIdx.x;            // 0..255
  int t = threadIdx.x;
  int b = r >> 7;
  const float scale = 0.125f;    // 1/sqrt(64)
  qs[t]       = qbuf[(size_t)r * D_ + t]       * scale;
  qs[t + 256] = qbuf[(size_t)r * D_ + 256 + t] * scale;
  if (t < TOPK) bidx[t] = idx[r * TOPK + t];
  __syncthreads();

  int h = t >> 5, lane = t & 31;
  float qreg[HD];
#pragma unroll
  for (int d = 0; d < HD; d++) qreg[d] = qs[h * HD + d];

  size_t base = (size_t)b * N_T * D_;
  float mymax = -INFINITY;
  for (int so = 0; so < 1024; so += 32) {
    int s  = so + lane;
    int gb = bidx[s >> 6];
    const __bf16* kr = Kf + base + (size_t)(gb * BS + (s & 63)) * D_ + h * HD;
    float acc = 0.f;
#pragma unroll
    for (int c = 0; c < 8; c++) {
      v8bf kv = *(const v8bf*)(kr + c * 8);
#pragma unroll
      for (int e = 0; e < 8; e++) acc += qreg[c * 8 + e] * (float)kv[e];
    }
    sc[h * 1024 + s] = acc;
    mymax = fmaxf(mymax, acc);
  }
#pragma unroll
  for (int off = 16; off; off >>= 1)
    mymax = fmaxf(mymax, __shfl_xor(mymax, off, 32));

  float sum = 0.f;
  for (int so = 0; so < 1024; so += 32) {
    int s = so + lane;
    float e = __expf(sc[h * 1024 + s] - mymax);
    sc[h * 1024 + s] = e;
    sum += e;
  }
#pragma unroll
  for (int off = 16; off; off >>= 1) sum += __shfl_xor(sum, off, 32);
  float inv = 1.f / sum;

  // wave wi consumed only its own head's scores -> no extra barrier needed
  float a0 = 0.f, a1 = 0.f;
  int d0 = lane * 2;
  for (int s = 0; s < 1024; s++) {
    float att = sc[h * 1024 + s];
    int gb = bidx[s >> 6];
    v2bf vv = *(const v2bf*)(Vf + base + (size_t)(gb * BS + (s & 63)) * D_ +
                             h * HD + d0);
    a0 += att * (float)vv[0];
    a1 += att * (float)vv[1];
  }
  obuf[(size_t)r * D_ + h * HD + d0]     = a0 * inv;
  obuf[(size_t)r * D_ + h * HD + d0 + 1] = a1 * inv;
}

// ---------------------------------------------------------------------------
extern "C" void kernel_launch(void* const* d_in, const int* in_sizes, int n_in,
                              void* d_out, int out_size, void* d_ws,
                              size_t ws_size, hipStream_t stream) {
  const float* latents = (const float*)d_in[0];
  const float* bytes   = (const float*)d_in[1];
  const float* rqw     = (const float*)d_in[2];
  const float* rqb     = (const float*)d_in[3];
  const float* rkw     = (const float*)d_in[4];
  const float* rkb     = (const float*)d_in[5];
  const float* Wq      = (const float*)d_in[6];
  const float* Wk      = (const float*)d_in[7];
  const float* Wv      = (const float*)d_in[8];
  const float* Wo      = (const float*)d_in[9];
  float* out = (float*)d_out;

  char* ws = (char*)d_ws;
  size_t off = 0;
  auto alloc = [&](size_t nbytes) -> void* {
    void* p = ws + off;
    off += (nbytes + 255) & ~(size_t)255;
    return p;
  };
  __bf16* Abuf = (__bf16*)alloc((size_t)MROWS * D_IN * 2);   // 16 MB
  __bf16* WkF  = (__bf16*)alloc((size_t)D_IN * D_ * 2);      // 256 KB
  __bf16* WvF  = (__bf16*)alloc((size_t)D_IN * D_ * 2);
  __bf16* Kf   = (__bf16*)alloc((size_t)MROWS * D_ * 2);     // 32 MB
  __bf16* Vf   = (__bf16*)alloc((size_t)MROWS * D_ * 2);     // 32 MB
  float* summ  = (float*)alloc((size_t)B_ * NB * D_IN * 4);
  float* rq    = (float*)alloc((size_t)NROWS * RDIM * 4);
  float* rk    = (float*)alloc((size_t)B_ * NB * RDIM * 4);
  float* lg    = (float*)alloc((size_t)NROWS * NB * 4);
  int*   idx   = (int*)alloc((size_t)NROWS * TOPK * 4);
  float* qbuf  = (float*)alloc((size_t)NROWS * D_ * 4);
  float* obuf  = (float*)alloc((size_t)NROWS * D_ * 4);

  // --- K/V precompute path (bf16 WMMA) ---
  nsca_prep_A<<<(MROWS / 16) * 8 * 32 / 256, 256, 0, stream>>>(bytes, Abuf);
  nsca_prep_B<<<32, 256, 0, stream>>>(Wk, WkF);
  nsca_prep_B<<<32, 256, 0, stream>>>(Wv, WvF);
  nsca_wmma_gemm<<<MROWS / 16, 256, 0, stream>>>(Abuf, WkF, Kf);
  nsca_wmma_gemm<<<MROWS / 16, 256, 0, stream>>>(Abuf, WvF, Vf);

  // --- router path ---
  nsca_summaries<<<B_ * NB, 256, 0, stream>>>(bytes, summ);
  nsca_sgemm<<<(NROWS * RDIM + 255) / 256, 256, 0, stream>>>(
      latents, rqw, rqb, rq, NROWS, RDIM, D_, 0);
  nsca_sgemm<<<(B_ * NB * RDIM + 255) / 256, 256, 0, stream>>>(
      summ, rkw, rkb, rk, B_ * NB, RDIM, D_IN, 0);
  for (int b = 0; b < B_; b++)
    nsca_sgemm<<<(N_L * NB + 255) / 256, 256, 0, stream>>>(
        rq + (size_t)b * N_L * RDIM, rk + (size_t)b * NB * RDIM, nullptr,
        lg + (size_t)b * N_L * NB, N_L, NB, RDIM, 1);
  nsca_topk<<<1, 256, 0, stream>>>(lg, idx);

  // --- attention path ---
  nsca_sgemm<<<(NROWS * D_ + 255) / 256, 256, 0, stream>>>(
      latents, Wq, nullptr, qbuf, NROWS, D_, D_, 0);
  nsca_attention<<<NROWS, 256, 0, stream>>>(qbuf, Kf, Vf, idx, obuf);
  nsca_sgemm<<<(NROWS * D_ + 255) / 256, 256, 0, stream>>>(
      obuf, Wo, nullptr, out, NROWS, D_, D_, 0);
}
